// VisionSdpaAttention_18691697672380
// MI455X (gfx1250) — matrix operand order repaired and checked by reading, not compiled, not run on hardware
//
#include <hip/hip_runtime.h>

// ---------------- types / helpers ----------------
typedef __bf16 bf16_t;
typedef __attribute__((ext_vector_type(16))) __bf16 v16bf;
typedef __attribute__((ext_vector_type(8)))  float  v8f;
typedef __attribute__((ext_vector_type(4)))  unsigned int v4u;

union Frag16 { v16bf v; v4u q[2]; };

#define L_  2048
#define B_  4
#define D_  1024
#define H_  16
#define HD_ 64
#define M_  (L_ * B_)      // 8192
#define N_QKV 3072

__device__ __forceinline__ bf16_t f2bf(float x) {
  union { float f; unsigned u; } a; a.f = x;
  unsigned r = a.u + 0x7FFFu + ((a.u >> 16) & 1u);
  union { unsigned short s; bf16_t b; } o; o.s = (unsigned short)(r >> 16);
  return o.b;
}
__device__ __forceinline__ float bf2f(bf16_t x) {
  union { unsigned short s; bf16_t b; } i; i.b = x;
  union { float f; unsigned u; } o; o.u = ((unsigned)i.s) << 16;
  return o.f;
}

// A-matrix fragment (16x32 bf16, MxK): lanes 0-15 rows M, K = {8g..8g+7, 16+8g..+7}
__device__ __forceinline__ Frag16 load_afrag(const bf16_t* base,
                                             int row0, int ld, int k0, int lane) {
  const int m = lane & 15, g = lane >> 4;
  const bf16_t* p = base + (size_t)(row0 + m) * ld + k0 + 8 * g;
  Frag16 f;
  f.q[0] = *(const v4u*)p;
  f.q[1] = *(const v4u*)(p + 16);
  return f;
}

// B-matrix fragment (32x16 bf16, KxN) from N-major storage bt[N][K]:
// lane (n + 16g) holds column n, K = {8g..8g+7, 16+8g..16+8g+7}
__device__ __forceinline__ Frag16 load_bfrag(const bf16_t* base,
                                             int n0, int ld, int k0, int lane) {
  const int n = lane & 15, g = lane >> 4;
  const bf16_t* p = base + (size_t)(n0 + n) * ld + k0 + 8 * g;
  Frag16 f;
  f.q[0] = *(const v4u*)p;
  f.q[1] = *(const v4u*)(p + 16);
  return f;
}

__device__ __forceinline__ v8f wmma_bf16(const Frag16& a, const Frag16& b, v8f c) {
  return __builtin_amdgcn_wmma_f32_16x16x32_bf16(false, a.v, false, b.v,
                                                 (short)0, c, false, false);
}

// CDNA5 async global->LDS copy (ASYNCcnt-tracked); 16 bytes per lane.
__device__ __forceinline__ void async_load_b128(unsigned lds_off, const void* gaddr) {
  asm volatile("global_load_async_to_lds_b128 %0, %1, off"
               :: "v"(lds_off), "v"((unsigned long long)(uintptr_t)gaddr)
               : "memory");
}
__device__ __forceinline__ void wait_asynccnt0() {
  asm volatile("s_wait_asynccnt 0x0" ::: "memory");
}

// ---------------- kernel 0: fp32 -> bf16 copy ----------------
__global__ void convert_kernel(const float* __restrict__ x, bf16_t* __restrict__ o, int n) {
  int i = blockIdx.x * blockDim.x + threadIdx.x;
  if (i < n) o[i] = f2bf(x[i]);
}

// ---------------- kernel 1: transpose W[K][N] -> Wt[N][K] (bf16) ----------------
__global__ void transpose_kernel(const float* __restrict__ w, bf16_t* __restrict__ o,
                                 int K, int N) {
  int i = blockIdx.x * blockDim.x + threadIdx.x;
  if (i < K * N) {
    int kk = i / N, nn = i - kk * N;
    o[(size_t)nn * K + kk] = f2bf(w[i]);
  }
}

// ---------------- kernel 2: qkv GEMM + scatter to q/k/v^T ----------------
__global__ __launch_bounds__(128) void qkv_gemm_kernel(
    const bf16_t* __restrict__ X,   // [M_, D_]
    const bf16_t* __restrict__ Wt,  // [N_QKV, D_]
    const float*  __restrict__ bias,// [N_QKV]
    bf16_t* __restrict__ qb,        // [B_][H_][L_][HD_]
    bf16_t* __restrict__ kb,        // [B_][H_][L_][HD_]
    bf16_t* __restrict__ vt)        // [B_][H_][HD_][L_]
{
  const int lane = threadIdx.x & 31;
  const int wave = threadIdx.x >> 5;
  const int tile = blockIdx.x * 4 + wave;
  const int NT = N_QKV / 64;            // 48
  const int tm = tile / NT, tn = tile - tm * NT;
  const int r0 = tm * 32, n0 = tn * 64;

  v8f acc[2][4];
#pragma unroll
  for (int i = 0; i < 2; ++i)
#pragma unroll
    for (int j = 0; j < 4; ++j) acc[i][j] = (v8f){0,0,0,0,0,0,0,0};

  for (int k0 = 0; k0 < D_; k0 += 32) {
    __builtin_prefetch(X + (size_t)(r0 + (lane & 15)) * D_ + k0 + 64, 0, 3);
    Frag16 a0 = load_afrag(X, r0,      D_, k0, lane);
    Frag16 a1 = load_afrag(X, r0 + 16, D_, k0, lane);
#pragma unroll
    for (int j = 0; j < 4; ++j) {
      Frag16 b = load_bfrag(Wt, n0 + 16 * j, D_, k0, lane);
      acc[0][j] = wmma_bf16(a0, b, acc[0][j]);
      acc[1][j] = wmma_bf16(a1, b, acc[1][j]);
    }
  }

  const int g = lane >> 4, cn = lane & 15;
  const int s = tn >> 4;          // 0=q 1=k 2=v
  const int h = tn & 15;
#pragma unroll
  for (int j = 0; j < 4; ++j) {
    const float bv = bias[n0 + 16 * j + cn];
    const int d = 16 * j + cn;
#pragma unroll
    for (int i = 0; i < 2; ++i) {
#pragma unroll
      for (int r = 0; r < 8; ++r) {
        int row = r0 + 16 * i + r + 8 * g;      // row = l*B + b
        int l = row >> 2, b = row & 3;
        float val = acc[i][j][r] + bv;
        if (s == 0)      qb[(((size_t)(b * H_ + h)) * L_ + l) * HD_ + d] = f2bf(val);
        else if (s == 1) kb[(((size_t)(b * H_ + h)) * L_ + l) * HD_ + d] = f2bf(val);
        else             vt[(((size_t)(b * H_ + h)) * HD_ + d) * L_ + l] = f2bf(val);
      }
    }
  }
}

// ---------------- kernel 3: RoPE on q (with 1/sqrt(Hd)) and k ----------------
__global__ void rope_kernel(const float* __restrict__ ang,
                            bf16_t* __restrict__ qb, bf16_t* __restrict__ kb) {
  int idx = blockIdx.x * blockDim.x + threadIdx.x;   // [0, B*H*L*32)
  int d = idx & 31;
  int l = (idx >> 5) & (L_ - 1);
  int bh = idx >> 16;
  float a = ang[l * 32 + d];
  float c = cosf(a), sn = sinf(a);
  size_t base = ((size_t)bh * L_ + l) * HD_ + d;
  const float scale = 0.125f;   // 1/sqrt(64)
  float q1 = bf2f(qb[base]), q2 = bf2f(qb[base + 32]);
  qb[base]      = f2bf((q1 * c - q2 * sn) * scale);
  qb[base + 32] = f2bf((q2 * c + q1 * sn) * scale);
  float k1 = bf2f(kb[base]), k2 = bf2f(kb[base + 32]);
  kb[base]      = f2bf(k1 * c - k2 * sn);
  kb[base + 32] = f2bf(k2 * c + k1 * sn);
}

// ---------------- kernel 4: flash attention with async-LDS K/V staging ----------------
__global__ __launch_bounds__(128) void attn_kernel(
    const bf16_t* __restrict__ qb,   // [B][H][L][64]  (scaled+roped)
    const bf16_t* __restrict__ kb,   // [B][H][L][64]  (roped)
    const bf16_t* __restrict__ vt,   // [B][H][64][L]
    bf16_t* __restrict__ outb)       // [L*B, D] bf16
{
  __shared__ __align__(16) bf16_t ldsK[2][32 * 64];   // 2 x 4KB, [key][hd]
  __shared__ __align__(16) bf16_t ldsV[2][64 * 32];   // 2 x 4KB, [hd][key]
  __shared__ __align__(16) bf16_t ldsP[4][16 * 32];   // per-wave P reshuffle

  const int t    = threadIdx.x;          // 0..127
  const int lane = t & 31;
  const int wave = t >> 5;
  const int qt = blockIdx.x & (L_ / 64 - 1);   // query tile of 64
  const int bh = blockIdx.x >> 5;              // b*H + h
  const int q0 = qt * 64 + wave * 16;
  const int g = lane >> 4, cn = lane & 15;

  const bf16_t* qbase = qb + (size_t)bh * L_ * HD_;
  const bf16_t* kbase = kb + (size_t)bh * L_ * HD_;
  const bf16_t* vbase = vt + (size_t)bh * HD_ * L_;

  // block-cooperative async stage of one 32-key chunk (K: 32x64, V: 64x32)
  auto stage = [&](int kc, int buf) {
    {
      int row = t >> 2, seg = t & 3;                 // K tile: 32B per thread
      const bf16_t* gp = kbase + (size_t)(kc + row) * HD_ + seg * 16;
      unsigned lo = (unsigned)(uintptr_t)(&ldsK[buf][row * 64 + seg * 16]);
      async_load_b128(lo, gp);
      async_load_b128(lo + 16, gp + 8);
    }
    {
      int row = t >> 1, seg = t & 1;                 // V tile: 32B per thread
      const bf16_t* gp = vbase + (size_t)row * L_ + kc + seg * 16;
      unsigned lo = (unsigned)(uintptr_t)(&ldsV[buf][row * 32 + seg * 16]);
      async_load_b128(lo, gp);
      async_load_b128(lo + 16, gp + 8);
    }
  };

  Frag16 aq0 = load_afrag(qbase, q0, HD_, 0, lane);
  Frag16 aq1 = load_afrag(qbase, q0, HD_, 32, lane);

  v8f o[4];
#pragma unroll
  for (int j = 0; j < 4; ++j) o[j] = (v8f){0,0,0,0,0,0,0,0};
  float m[8], lsum[8];
#pragma unroll
  for (int r = 0; r < 8; ++r) { m[r] = -1e30f; lsum[r] = 0.f; }

  bf16_t* pbase = &ldsP[wave][0];

  stage(0, 0);                                  // prologue

  for (int kc = 0; kc < L_; kc += 32) {
    const int cur = (kc >> 5) & 1;
    wait_asynccnt0();            // my staged loads for `cur` are done
    __syncthreads();             // everyone's loads done + prev compute done
    stage((kc + 32) & (L_ - 1), cur ^ 1);       // overlap next DMA with compute

    const bf16_t* Kt = &ldsK[cur][0];
    const bf16_t* Vt = &ldsV[cur][0];

    // S = Q K^T for 16 queries x 32 keys (contraction over head dim 64)
    v8f s0 = (v8f){0,0,0,0,0,0,0,0}, s1 = s0;
    {
      Frag16 b00 = load_bfrag(Kt, 0,  64, 0,  lane);
      Frag16 b01 = load_bfrag(Kt, 0,  64, 32, lane);
      s0 = wmma_bf16(aq0, b00, s0);
      s0 = wmma_bf16(aq1, b01, s0);
      Frag16 b10 = load_bfrag(Kt, 16, 64, 0,  lane);
      Frag16 b11 = load_bfrag(Kt, 16, 64, 32, lane);
      s1 = wmma_bf16(aq0, b10, s1);
      s1 = wmma_bf16(aq1, b11, s1);
    }
    // online softmax (rows r+8g; reduce across the 16 lanes of each half)
#pragma unroll
    for (int r = 0; r < 8; ++r) {
      float mx = fmaxf(s0[r], s1[r]);
      mx = fmaxf(mx, __shfl_xor(mx, 1, 32));
      mx = fmaxf(mx, __shfl_xor(mx, 2, 32));
      mx = fmaxf(mx, __shfl_xor(mx, 4, 32));
      mx = fmaxf(mx, __shfl_xor(mx, 8, 32));
      float mn = fmaxf(m[r], mx);
      float fac = __expf(m[r] - mn);
      m[r] = mn;
      float p0 = __expf(s0[r] - mn);
      float p1 = __expf(s1[r] - mn);
      s0[r] = p0; s1[r] = p1;
      float sum = p0 + p1;
      sum += __shfl_xor(sum, 1, 32);
      sum += __shfl_xor(sum, 2, 32);
      sum += __shfl_xor(sum, 4, 32);
      sum += __shfl_xor(sum, 8, 32);
      lsum[r] = lsum[r] * fac + sum;
      o[0][r] *= fac; o[1][r] *= fac; o[2][r] *= fac; o[3][r] *= fac;
    }
    // reshape P (C-layout) -> A-layout via per-wave LDS tile (16 rows x 32 cols)
    asm volatile("" ::: "memory");
    __builtin_amdgcn_wave_barrier();
#pragma unroll
    for (int r = 0; r < 8; ++r) {
      pbase[(r + 8 * g) * 32 + cn]      = f2bf(s0[r]);
      pbase[(r + 8 * g) * 32 + 16 + cn] = f2bf(s1[r]);
    }
    asm volatile("" ::: "memory");
    __builtin_amdgcn_wave_barrier();
    Frag16 pf;
    {
      const bf16_t* pp = pbase + (lane & 15) * 32 + 8 * g;
      pf.q[0] = *(const v4u*)pp;
      pf.q[1] = *(const v4u*)(pp + 16);
    }
    asm volatile("" ::: "memory");
    __builtin_amdgcn_wave_barrier();
    // O += P @ V   (contraction over 32 keys; N = head dim)
#pragma unroll
    for (int j = 0; j < 4; ++j) {
      Frag16 vf = load_bfrag(Vt, 16 * j, 32, 0, lane);
      o[j] = wmma_bf16(pf, vf, o[j]);
    }
  }

  // finalize: out[(q*B + b)*D + h*64 + d]
  const int b = bh >> 4, h = bh & 15;
#pragma unroll
  for (int j = 0; j < 4; ++j) {
#pragma unroll
    for (int r = 0; r < 8; ++r) {
      int qrow = q0 + r + 8 * g;
      float val = o[j][r] / lsum[r];
      outb[((size_t)qrow * B_ + b) * D_ + h * HD_ + 16 * j + cn] = f2bf(val);
    }
  }
}

// ---------------- kernel 5: out-proj GEMM (bf16 x bf16 -> f32) ----------------
__global__ __launch_bounds__(128) void out_gemm_kernel(
    const bf16_t* __restrict__ A,    // [M_, D_]
    const bf16_t* __restrict__ Wt,   // [D_, D_]  (N-major)
    const float*  __restrict__ bias, // [D_]
    float* __restrict__ out)         // [M_, D_]
{
  const int lane = threadIdx.x & 31;
  const int wave = threadIdx.x >> 5;
  const int tile = blockIdx.x * 4 + wave;
  const int tm = tile >> 4, tn = tile & 15;
  const int r0 = tm * 32, n0 = tn * 64;

  v8f acc[2][4];
#pragma unroll
  for (int i = 0; i < 2; ++i)
#pragma unroll
    for (int j = 0; j < 4; ++j) acc[i][j] = (v8f){0,0,0,0,0,0,0,0};

  for (int k0 = 0; k0 < D_; k0 += 32) {
    __builtin_prefetch(A + (size_t)(r0 + (lane & 15)) * D_ + k0 + 64, 0, 3);
    Frag16 a0 = load_afrag(A, r0,      D_, k0, lane);
    Frag16 a1 = load_afrag(A, r0 + 16, D_, k0, lane);
#pragma unroll
    for (int j = 0; j < 4; ++j) {
      Frag16 b = load_bfrag(Wt, n0 + 16 * j, D_, k0, lane);
      acc[0][j] = wmma_bf16(a0, b, acc[0][j]);
      acc[1][j] = wmma_bf16(a1, b, acc[1][j]);
    }
  }

  const int g = lane >> 4, cn = lane & 15;
#pragma unroll
  for (int j = 0; j < 4; ++j) {
    const int col = n0 + 16 * j + cn;
    const float bv = bias[col];
#pragma unroll
    for (int i = 0; i < 2; ++i) {
#pragma unroll
      for (int r = 0; r < 8; ++r) {
        int row = r0 + 16 * i + r + 8 * g;
        out[(size_t)row * D_ + col] = acc[i][j][r] + bv;
      }
    }
  }
}

// ---------------- launch ----------------
extern "C" void kernel_launch(void* const* d_in, const int* in_sizes, int n_in,
                              void* d_out, int out_size, void* d_ws, size_t ws_size,
                              hipStream_t stream) {
  (void)in_sizes; (void)n_in; (void)out_size; (void)ws_size;
  const float* hidden = (const float*)d_in[0];
  const float* angles = (const float*)d_in[1];
  const float* Wqkv   = (const float*)d_in[2];
  const float* bqkv   = (const float*)d_in[3];
  const float* Wout   = (const float*)d_in[4];
  const float* bout   = (const float*)d_in[5];
  float* out = (float*)d_out;

  char* ws = (char*)d_ws;
  const size_t SZ_X   = (size_t)M_ * D_ * 2;        // 16 MB
  const size_t SZ_WQ  = (size_t)N_QKV * D_ * 2;     //  6 MB
  const size_t SZ_WO  = (size_t)D_ * D_ * 2;        //  2 MB
  const size_t SZ_QKV = (size_t)B_ * H_ * L_ * HD_ * 2; // 16 MB each
  bf16_t* Xbf   = (bf16_t*)(ws);
  bf16_t* WqkvT = (bf16_t*)(ws + SZ_X);
  bf16_t* WoutT = (bf16_t*)(ws + SZ_X + SZ_WQ);
  bf16_t* qbuf  = (bf16_t*)(ws + SZ_X + SZ_WQ + SZ_WO);
  bf16_t* kbuf  = (bf16_t*)(ws + SZ_X + SZ_WQ + SZ_WO + SZ_QKV);
  bf16_t* vtbuf = (bf16_t*)(ws + SZ_X + SZ_WQ + SZ_WO + 2 * SZ_QKV);
  bf16_t* attnb = (bf16_t*)(ws + SZ_X + SZ_WQ + SZ_WO + 3 * SZ_QKV);

  convert_kernel<<<(M_ * D_) / 256, 256, 0, stream>>>(hidden, Xbf, M_ * D_);
  transpose_kernel<<<(D_ * N_QKV) / 256, 256, 0, stream>>>(Wqkv, WqkvT, D_, N_QKV);
  transpose_kernel<<<(D_ * D_) / 256, 256, 0, stream>>>(Wout, WoutT, D_, D_);

  // 8192/32 * 3072/64 = 12288 wave-tiles, 4 waves per block
  qkv_gemm_kernel<<<12288 / 4, 128, 0, stream>>>(Xbf, WqkvT, bqkv, qbuf, kbuf, vtbuf);

  rope_kernel<<<(B_ * H_ * L_ * 32) / 256, 256, 0, stream>>>(angles, qbuf, kbuf);

  // B*H * L/64 blocks, 4 waves (16 queries each)
  attn_kernel<<<B_ * H_ * (L_ / 64), 128, 0, stream>>>(qbuf, kbuf, vtbuf, attnb);

  // 8192/32 * 1024/64 = 4096 wave-tiles
  out_gemm_kernel<<<4096 / 4, 128, 0, stream>>>(attnb, WoutT, bout, out);
}
